// Block_84310208020549
// MI455X (gfx1250) — matrix operand order repaired and checked by reading, not compiled, not run on hardware
//
#include <hip/hip_runtime.h>
#include <hip/hip_bf16.h>
#include <math.h>

typedef __bf16 bf16;
typedef __bf16 v16bf  __attribute__((ext_vector_type(16)));
typedef __bf16 bf16x8 __attribute__((ext_vector_type(8)));
typedef __bf16 bf16x4 __attribute__((ext_vector_type(4)));
typedef float  v8f    __attribute__((ext_vector_type(8)));
typedef int    v4i    __attribute__((ext_vector_type(4)));

typedef __attribute__((address_space(1))) v4i* gptr4;   // global int4*
typedef __attribute__((address_space(3))) v4i* lptr4;   // LDS int4*

#define WMMA_BF16(a,b,c) \
  __builtin_amdgcn_wmma_f32_16x16x32_bf16(false,(a),false,(b),(short)0,(c),false,false)

#if defined(__has_builtin)
#if __has_builtin(__builtin_amdgcn_global_load_async_to_lds_b128) && \
    __has_builtin(__builtin_amdgcn_s_wait_asynccnt)
#define HAVE_ASYNC_LDS 1
#endif
#endif

// copy 16B global -> LDS (async-to-LDS path when available)
__device__ __forceinline__ void stage16B(const void* g, void* l)
{
#ifdef HAVE_ASYNC_LDS
  __builtin_amdgcn_global_load_async_to_lds_b128(
      (gptr4)(unsigned long long)g,
      (lptr4)(unsigned)(unsigned long long)l,
      0, 0);
#else
  *(bf16x8*)l = *(const bf16x8*)g;
#endif
}
__device__ __forceinline__ void stage_wait()
{
#ifdef HAVE_ASYNC_LDS
  __builtin_amdgcn_s_wait_asynccnt(0);
#endif
}

static constexpr int BATCH = 2;
static constexpr int T     = 2048;
static constexpr int DIM   = 1024;
static constexpr int HEADS = 16;
static constexpr int HD    = 64;
static constexpr int NE    = 8;      // experts
static constexpr int TOPK  = 2;
static constexpr int HID   = 4096;
static constexpr int NTOK  = BATCH * T;      // 4096
static constexpr int NSLOT = NTOK * TOPK;    // 8192

// ---------------------------------------------------------------- LayerNorm 1
__global__ __launch_bounds__(256)
void ln_kernel(const float* __restrict__ x, const float* __restrict__ g,
               const float* __restrict__ bb, bf16* __restrict__ h)
{
  const int n = blockIdx.x;
  const int tid = threadIdx.x;
  const float* xr = x + (size_t)n * DIM;
  float4 v = ((const float4*)xr)[tid];

  __shared__ float red[256];
  red[tid] = v.x + v.y + v.z + v.w;
  __syncthreads();
  for (int st = 128; st > 0; st >>= 1) { if (tid < st) red[tid] += red[tid + st]; __syncthreads(); }
  const float mu = red[0] * (1.0f / DIM);
  __syncthreads();
  const float dx = v.x - mu, dy = v.y - mu, dz = v.z - mu, dw = v.w - mu;
  red[tid] = dx*dx + dy*dy + dz*dz + dw*dw;
  __syncthreads();
  for (int st = 128; st > 0; st >>= 1) { if (tid < st) red[tid] += red[tid + st]; __syncthreads(); }
  const float rstd = rsqrtf(red[0] * (1.0f / DIM) + 1e-5f);

  const int d0 = tid * 4;
  bf16* hr = h + (size_t)n * DIM + d0;
  hr[0] = (bf16)(dx * rstd * g[d0 + 0] + bb[d0 + 0]);
  hr[1] = (bf16)(dy * rstd * g[d0 + 1] + bb[d0 + 1]);
  hr[2] = (bf16)(dz * rstd * g[d0 + 2] + bb[d0 + 2]);
  hr[3] = (bf16)(dw * rstd * g[d0 + 3] + bb[d0 + 3]);
}

// --------------------------------------------------- LayerNorm 2 + router top2
__global__ __launch_bounds__(256)
void ln2_router_kernel(const float* __restrict__ x2, const float* __restrict__ g,
                       const float* __restrict__ bb, const float* __restrict__ wr,
                       bf16* __restrict__ h2, float* __restrict__ topw, int* __restrict__ topi)
{
  const int n = blockIdx.x;
  const int tid = threadIdx.x;
  const float* xr = x2 + (size_t)n * DIM;
  float4 v = ((const float4*)xr)[tid];

  __shared__ float red[256];
  __shared__ float lsh[NE][256];

  red[tid] = v.x + v.y + v.z + v.w;
  __syncthreads();
  for (int st = 128; st > 0; st >>= 1) { if (tid < st) red[tid] += red[tid + st]; __syncthreads(); }
  const float mu = red[0] * (1.0f / DIM);
  __syncthreads();
  const float d0f = v.x - mu, d1f = v.y - mu, d2f = v.z - mu, d3f = v.w - mu;
  red[tid] = d0f*d0f + d1f*d1f + d2f*d2f + d3f*d3f;
  __syncthreads();
  for (int st = 128; st > 0; st >>= 1) { if (tid < st) red[tid] += red[tid + st]; __syncthreads(); }
  const float rstd = rsqrtf(red[0] * (1.0f / DIM) + 1e-5f);

  const int d0 = tid * 4;
  float nv[4];
  nv[0] = d0f * rstd * g[d0 + 0] + bb[d0 + 0];
  nv[1] = d1f * rstd * g[d0 + 1] + bb[d0 + 1];
  nv[2] = d2f * rstd * g[d0 + 2] + bb[d0 + 2];
  nv[3] = d3f * rstd * g[d0 + 3] + bb[d0 + 3];

  bf16* hr = h2 + (size_t)n * DIM + d0;
  hr[0] = (bf16)nv[0]; hr[1] = (bf16)nv[1]; hr[2] = (bf16)nv[2]; hr[3] = (bf16)nv[3];

  float part[NE];
  #pragma unroll
  for (int e = 0; e < NE; ++e) part[e] = 0.0f;
  #pragma unroll
  for (int k = 0; k < 4; ++k) {
    const float* wrow = wr + (size_t)(d0 + k) * NE;
    const float t = nv[k];
    #pragma unroll
    for (int e = 0; e < NE; ++e) part[e] += t * wrow[e];
  }
  #pragma unroll
  for (int e = 0; e < NE; ++e) lsh[e][tid] = part[e];
  __syncthreads();
  if (tid < NE) {                 // deterministic serial reduce per expert
    float s = 0.0f;
    for (int i = 0; i < 256; ++i) s += lsh[tid][i];
    lsh[tid][0] = s;
  }
  __syncthreads();
  if (tid == 0) {
    float lg[NE];
    #pragma unroll
    for (int e = 0; e < NE; ++e) lg[e] = lsh[e][0];
    int i1 = 0;
    for (int e = 1; e < NE; ++e) if (lg[e] > lg[i1]) i1 = e;
    int i2 = -1;
    for (int e = 0; e < NE; ++e) if (e != i1 && (i2 < 0 || lg[e] > lg[i2])) i2 = e;
    const float e2 = __expf(lg[i2] - lg[i1]);
    const float inv = 1.0f / (1.0f + e2);
    topw[n*2+0] = inv;        topw[n*2+1] = e2 * inv;
    topi[n*2+0] = i1;         topi[n*2+1] = i2;
  }
}

// ------------------------------------------------------------------ bf16 GEMM
// block tile 128x128, 8 waves, wave tile 32x64 (2x4 WMMA 16x16x32)
enum { EPI_BF16 = 0, EPI_RESID = 1, EPI_GELU = 2, EPI_SCALE = 3 };

template<int EPI>
__global__ __launch_bounds__(256)
void gemm_bf16_kernel(const bf16* __restrict__ A, const float* __restrict__ W,
                      const float* __restrict__ bias, const float* __restrict__ resid,
                      float* __restrict__ outF, bf16* __restrict__ outB,
                      const float* __restrict__ rowScale,
                      const int* __restrict__ segOff, const int* __restrict__ segCnt,
                      int M, int Kd, int Nc)
{
  const int z    = blockIdx.z;
  const int base = segOff ? segOff[z] : 0;
  const int rows = segCnt ? segCnt[z] : M;
  const int bm = blockIdx.y * 128;
  if (bm >= rows) return;
  const int bn = blockIdx.x * 128;

  const bf16*  Ab = A + (size_t)base * Kd;
  const float* Wz = W + (size_t)z * (size_t)Kd * Nc;

  __shared__ bf16 As [128][40];   // 128 rows x 32 k, padded (80B row, 16B aligned)
  __shared__ bf16 BsT[128][40];   // TRANSPOSED: 128 cols x 32 k, padded

  const int tid  = threadIdx.x;
  const int wave = tid >> 5, lane = tid & 31;
  const int wm = wave >> 1, wn = wave & 1;     // 4 x 2 wave grid
  const int r15 = lane & 15, hi = lane >> 4;
  const int kb8 = hi * 8, kb16 = hi * 16;

  const int arow = tid >> 1, acb = (tid & 1) * 16;      // A: 2 thr/row, 16 elems
  const int brow = tid >> 3, bcb = (tid & 7) * 16;      // B: 8 thr/row, 16 cols

  v8f acc[2][4];
  #pragma unroll
  for (int mt = 0; mt < 2; ++mt)
    #pragma unroll
    for (int nt = 0; nt < 4; ++nt)
      #pragma unroll
      for (int i = 0; i < 8; ++i) acc[mt][nt][i] = 0.0f;

  for (int k0 = 0; k0 < Kd; k0 += 32) {
    // ---- stage A tile (bf16 copy; async-to-LDS when available)
    if (bm + arow < rows) {
      const bf16* src = Ab + (size_t)(bm + arow) * Kd + k0 + acb;
      stage16B(src,     &As[arow][acb]);
      stage16B(src + 8, &As[arow][acb + 8]);
    } else {
      #pragma unroll
      for (int i = 0; i < 16; ++i) As[arow][acb + i] = (bf16)0.0f;
    }
    // ---- stage B tile transposed with f32->bf16 conversion
    {
      const float* bsrc = Wz + (size_t)(k0 + brow) * Nc + bn + bcb;
      if (k0 + 32 < Kd) __builtin_prefetch(bsrc + (size_t)32 * Nc, 0, 1);
      float fv[16];
      #pragma unroll
      for (int q = 0; q < 4; ++q) {
        float4 f = ((const float4*)bsrc)[q];
        fv[q*4+0] = f.x; fv[q*4+1] = f.y; fv[q*4+2] = f.z; fv[q*4+3] = f.w;
      }
      #pragma unroll
      for (int j = 0; j < 16; ++j) BsT[bcb + j][brow] = (bf16)fv[j];
    }
    stage_wait();
    __syncthreads();

    // ---- fragments: all contiguous 16B LDS reads
    v16bf af[2], bfv[4];
    #pragma unroll
    for (int mt = 0; mt < 2; ++mt) {
      const int r = wm * 32 + mt * 16 + r15;
      bf16x8 lo = *(const bf16x8*)&As[r][kb8];
      bf16x8 hb = *(const bf16x8*)&As[r][kb8 + 16];
      #pragma unroll
      for (int i = 0; i < 8; ++i) { af[mt][i] = lo[i]; af[mt][i + 8] = hb[i]; }
    }
    #pragma unroll
    for (int nt = 0; nt < 4; ++nt) {
      const int c = wn * 64 + nt * 16 + r15;
      bf16x8 lo = *(const bf16x8*)&BsT[c][kb8];
      bf16x8 hb = *(const bf16x8*)&BsT[c][kb8 + 16];
      #pragma unroll
      for (int i = 0; i < 8; ++i) { bfv[nt][i] = lo[i]; bfv[nt][i + 8] = hb[i]; }
    }
    #pragma unroll
    for (int mt = 0; mt < 2; ++mt)
      #pragma unroll
      for (int nt = 0; nt < 4; ++nt)
        acc[mt][nt] = WMMA_BF16(af[mt], bfv[nt], acc[mt][nt]);
    __syncthreads();
  }

  #pragma unroll
  for (int mt = 0; mt < 2; ++mt)
    #pragma unroll
    for (int nt = 0; nt < 4; ++nt)
      #pragma unroll
      for (int v = 0; v < 8; ++v) {
        const int m = bm + wm * 32 + mt * 16 + hi * 8 + v;
        if (m >= rows) continue;
        const int c = bn + wn * 64 + nt * 16 + r15;
        const float val = acc[mt][nt][v];
        const size_t gi = (size_t)(base + m) * Nc + c;
        if (EPI == EPI_BF16) {
          outB[gi] = (bf16)val;
        } else if (EPI == EPI_RESID) {
          outF[gi] = resid[gi] + val;
        } else if (EPI == EPI_GELU) {
          const float t = val + bias[(size_t)z * Nc + c];
          outB[gi] = (bf16)(0.5f * t * (1.0f + erff(t * 0.70710678118f)));
        } else {
          outF[gi] = (val + bias[(size_t)z * Nc + c]) * rowScale[base + m];
        }
      }
}

// ------------------------------------------------------------ flash attention
__global__ __launch_bounds__(256)
void attn_kernel(const bf16* __restrict__ qkv, bf16* __restrict__ y)
{
  const int qt = blockIdx.x;        // query tile of 128
  const int h  = blockIdx.y;
  const int bb = blockIdx.z;
  const int tid = threadIdx.x, wave = tid >> 5, lane = tid & 31;
  const int r15 = lane & 15, hi = lane >> 4;
  const int kb8 = hi * 8, kb16 = hi * 16;

  __shared__ bf16  Ks [64][72];     // keys x hd   (row-major, contiguous frags)
  __shared__ bf16  VsT[64][72];     // TRANSPOSED: hd x keys (contiguous frags)
  __shared__ float Ss[8][16][64];   // S then P strip per wave
  __shared__ float rowA[8][16];

  const size_t tokBase = (size_t)bb * T;
  const int qrow0 = qt * 128 + wave * 16;

  // Q fragments (A-layout), loaded once from global
  v16bf qa[2];
  {
    const bf16* qp = qkv + (tokBase + qrow0 + r15) * (size_t)(3 * DIM) + h * HD;
    #pragma unroll
    for (int ks = 0; ks < 2; ++ks) {
      bf16x8 lo = *(const bf16x8*)(qp + ks * 32 + kb8);
      bf16x8 hb = *(const bf16x8*)(qp + ks * 32 + kb8 + 16);
      #pragma unroll
      for (int i = 0; i < 8; ++i) { qa[ks][i] = lo[i]; qa[ks][8 + i] = hb[i]; }
    }
  }

  v8f oacc[4];
  #pragma unroll
  for (int nt = 0; nt < 4; ++nt)
    #pragma unroll
    for (int i = 0; i < 8; ++i) oacc[nt][i] = 0.0f;

  float mrow = -3.0e38f, lrow = 0.0f;   // meaningful for lanes 0..15

  const int nkt = 2 * qt + 2;
  const int krow = tid >> 2, kcb = (tid & 3) * 16;

  for (int kt = 0; kt < nkt; ++kt) {
    __syncthreads();
    {
      const size_t rp = (tokBase + (size_t)kt * 64 + krow) * (size_t)(3 * DIM) + h * HD;
      // K tile: straight copy (async path)
      stage16B(qkv + rp + DIM + kcb,     &Ks[krow][kcb]);
      stage16B(qkv + rp + DIM + kcb + 8, &Ks[krow][kcb + 8]);
      // V tile: transpose while staging
      bf16x8 v0 = *(const bf16x8*)(qkv + rp + 2 * DIM + kcb);
      bf16x8 v1 = *(const bf16x8*)(qkv + rp + 2 * DIM + kcb + 8);
      #pragma unroll
      for (int j = 0; j < 8; ++j) {
        VsT[kcb + j][krow]     = v0[j];
        VsT[kcb + 8 + j][krow] = v1[j];
      }
    }
    stage_wait();
    __syncthreads();

    const bool active = (kt * 64) <= (qrow0 + 15);

    if (active) {                              // S = Q * K^T
      v8f s[4];
      #pragma unroll
      for (int nt = 0; nt < 4; ++nt)
        #pragma unroll
        for (int i = 0; i < 8; ++i) s[nt][i] = 0.0f;
      #pragma unroll
      for (int nt = 0; nt < 4; ++nt) {
        const int c = nt * 16 + r15;           // key column
        #pragma unroll
        for (int ks = 0; ks < 2; ++ks) {
          v16bf kf;
          bf16x8 lo = *(const bf16x8*)&Ks[c][ks * 32 + kb8];
          bf16x8 hb = *(const bf16x8*)&Ks[c][ks * 32 + kb8 + 16];
          #pragma unroll
          for (int i = 0; i < 8; ++i) { kf[i] = lo[i]; kf[i + 8] = hb[i]; }
          s[nt] = WMMA_BF16(qa[ks], kf, s[nt]);
        }
      }
      #pragma unroll
      for (int nt = 0; nt < 4; ++nt)
        #pragma unroll
        for (int v = 0; v < 8; ++v)
          Ss[wave][v + 8 * hi][nt * 16 + r15] = s[nt][v];
    }
    __syncthreads();

    if (active && lane < 16) {                 // online softmax, lanes own rows
      const int qg = qrow0 + lane;
      const float mprev = mrow;
      float rmax = mprev;
      for (int c = 0; c < 64; ++c) {
        float sv = Ss[wave][lane][c] * 0.125f;
        if (kt * 64 + c > qg) sv = -3.0e38f;
        rmax = fmaxf(rmax, sv);
      }
      const float mnew = rmax;
      const float resc = __expf(mprev - mnew);
      float lsum = lrow * resc;
      for (int c = 0; c < 64; ++c) {
        const float sv = Ss[wave][lane][c] * 0.125f;
        const float p = (kt * 64 + c > qg) ? 0.0f : __expf(sv - mnew);
        lsum += p;
        Ss[wave][lane][c] = p;                 // reuse strip as P
      }
      mrow = mnew; lrow = lsum; rowA[wave][lane] = resc;
    }
    __syncthreads();

    if (active) {                              // O = O*resc + P*V
      #pragma unroll
      for (int nt = 0; nt < 4; ++nt)
        #pragma unroll
        for (int v = 0; v < 8; ++v) oacc[nt][v] *= rowA[wave][v + 8 * hi];

      v16bf pf[2];
      #pragma unroll
      for (int ks = 0; ks < 2; ++ks)
        #pragma unroll
        for (int i = 0; i < 8; ++i) {
          pf[ks][i]     = (bf16)Ss[wave][r15][ks * 32 + kb8 + i];
          pf[ks][i + 8] = (bf16)Ss[wave][r15][ks * 32 + kb8 + 16 + i];
        }
      #pragma unroll
      for (int nt = 0; nt < 4; ++nt) {
        const int c = nt * 16 + r15;           // hd column
        #pragma unroll
        for (int ks = 0; ks < 2; ++ks) {
          v16bf vf;
          bf16x8 lo = *(const bf16x8*)&VsT[c][ks * 32 + kb8];
          bf16x8 hb = *(const bf16x8*)&VsT[c][ks * 32 + kb8 + 16];
          #pragma unroll
          for (int i = 0; i < 8; ++i) { vf[i] = lo[i]; vf[i + 8] = hb[i]; }
          oacc[nt] = WMMA_BF16(pf[ks], vf, oacc[nt]);
        }
      }
    }
  }

  __syncthreads();
  if (lane < 16) rowA[wave][lane] = 1.0f / lrow;
  __syncthreads();
  #pragma unroll
  for (int nt = 0; nt < 4; ++nt)
    #pragma unroll
    for (int v = 0; v < 8; ++v) {
      const int r = v + 8 * hi;
      const float o = oacc[nt][v] * rowA[wave][r];
      y[(tokBase + qrow0 + r) * (size_t)DIM + h * HD + nt * 16 + r15] = (bf16)o;
    }
}

// ------------------------------------------------------------- MoE plumbing
__global__ void moe_init_kernel(int* counts, int* cursor)
{
  const int t = threadIdx.x;
  if (t < NE) { counts[t] = 0; cursor[t] = 0; }
}

__global__ void moe_count_kernel(const int* __restrict__ topi, int* __restrict__ counts)
{
  const int idx = blockIdx.x * 256 + threadIdx.x;
  if (idx < NSLOT) atomicAdd(&counts[topi[idx]], 1);
}

__global__ void moe_offsets_kernel(const int* __restrict__ counts, int* __restrict__ offs)
{
  if (threadIdx.x == 0) {
    int s = 0;
    for (int e = 0; e < NE; ++e) { offs[e] = s; s += counts[e]; }
  }
}

__global__ void moe_assign_kernel(const int* __restrict__ topi, const float* __restrict__ topw,
                                  const int* __restrict__ offs, int* __restrict__ cursor,
                                  int* __restrict__ rowOf, float* __restrict__ wtOf,
                                  int* __restrict__ pos)
{
  const int idx = blockIdx.x * 256 + threadIdx.x;
  if (idx >= NSLOT) return;
  const int e = topi[idx];
  const int slot = offs[e] + atomicAdd(&cursor[e], 1);
  rowOf[slot] = idx >> 1;
  wtOf[slot]  = topw[idx];
  pos[idx]    = slot;
}

__global__ __launch_bounds__(256)
void moe_gather_kernel(const int* __restrict__ rowOf, const bf16* __restrict__ h2,
                       bf16* __restrict__ hg)
{
  const int slot = blockIdx.x;
  const int n = rowOf[slot];
  ((bf16x4*)(hg + (size_t)slot * DIM))[threadIdx.x] =
      ((const bf16x4*)(h2 + (size_t)n * DIM))[threadIdx.x];
}

__global__ __launch_bounds__(256)
void final_combine_kernel(const float* __restrict__ x2, const float* __restrict__ outg,
                          const int* __restrict__ pos, float* __restrict__ out)
{
  const int n = blockIdx.x;
  const int c = blockIdx.y * 256 + threadIdx.x;
  const size_t gi = (size_t)n * DIM + c;
  const int s0 = pos[n * 2 + 0], s1 = pos[n * 2 + 1];
  out[gi] = x2[gi] + outg[(size_t)s0 * DIM + c] + outg[(size_t)s1 * DIM + c];
}

// ---------------------------------------------------------------- launcher
extern "C" void kernel_launch(void* const* d_in, const int* in_sizes, int n_in,
                              void* d_out, int out_size, void* d_ws, size_t ws_size,
                              hipStream_t stream)
{
  const float* x      = (const float*)d_in[0];
  const float* ln1_g  = (const float*)d_in[1];
  const float* ln1_b  = (const float*)d_in[2];
  const float* w_attn = (const float*)d_in[3];
  const float* w_proj = (const float*)d_in[4];
  const float* ln2_g  = (const float*)d_in[5];
  const float* ln2_b  = (const float*)d_in[6];
  const float* w_rout = (const float*)d_in[7];
  const float* w1     = (const float*)d_in[8];
  const float* b1     = (const float*)d_in[9];
  const float* w2     = (const float*)d_in[10];
  const float* b2     = (const float*)d_in[11];
  float* out = (float*)d_out;

  char* w = (char*)d_ws;
  size_t off = 0;
  auto carve = [&](size_t bytes) -> void* {
    void* p = w + off;
    off = (off + bytes + 255) & ~(size_t)255;
    return p;
  };
  bf16*  h1    = (bf16*) carve((size_t)NTOK * DIM * 2);
  bf16*  qkv   = (bf16*) carve((size_t)NTOK * 3 * DIM * 2);
  bf16*  yb    = (bf16*) carve((size_t)NTOK * DIM * 2);
  float* x2    = (float*)carve((size_t)NTOK * DIM * 4);
  bf16*  h2    = (bf16*) carve((size_t)NTOK * DIM * 2);
  float* topw  = (float*)carve((size_t)NSLOT * 4);
  int*   topi  = (int*)  carve((size_t)NSLOT * 4);
  int*   counts= (int*)  carve(256);
  int*   cursor= (int*)  carve(256);
  int*   offs  = (int*)  carve(256);
  int*   rowOf = (int*)  carve((size_t)NSLOT * 4);
  float* wtOf  = (float*)carve((size_t)NSLOT * 4);
  int*   pos   = (int*)  carve((size_t)NSLOT * 4);
  bf16*  hg    = (bf16*) carve((size_t)NSLOT * DIM * 2);
  bf16*  he    = (bf16*) carve((size_t)NSLOT * HID * 2);
  float* outg  = (float*)carve((size_t)NSLOT * DIM * 4);
  (void)ws_size; (void)in_sizes; (void)n_in; (void)out_size;

  // 1. LN1 -> bf16
  ln_kernel<<<NTOK, 256, 0, stream>>>(x, ln1_g, ln1_b, h1);

  // 2. QKV GEMM: [4096,1024] x [1024,3072] -> bf16
  gemm_bf16_kernel<EPI_BF16><<<dim3(3 * DIM / 128, NTOK / 128, 1), 256, 0, stream>>>(
      h1, w_attn, nullptr, nullptr, nullptr, qkv, nullptr, nullptr, nullptr,
      NTOK, DIM, 3 * DIM);

  // 3. causal flash attention
  attn_kernel<<<dim3(T / 128, HEADS, BATCH), 256, 0, stream>>>(qkv, yb);

  // 4. output projection + residual: x2 = x + y @ w_proj
  gemm_bf16_kernel<EPI_RESID><<<dim3(DIM / 128, NTOK / 128, 1), 256, 0, stream>>>(
      yb, w_proj, nullptr, x, x2, nullptr, nullptr, nullptr, nullptr,
      NTOK, DIM, DIM);

  // 5. LN2 + router top-2
  ln2_router_kernel<<<NTOK, 256, 0, stream>>>(x2, ln2_g, ln2_b, w_rout, h2, topw, topi);

  // 6. routing: count -> offsets -> assign -> gather
  moe_init_kernel<<<1, 32, 0, stream>>>(counts, cursor);
  moe_count_kernel<<<NSLOT / 256, 256, 0, stream>>>(topi, counts);
  moe_offsets_kernel<<<1, 32, 0, stream>>>(counts, offs);
  moe_assign_kernel<<<NSLOT / 256, 256, 0, stream>>>(topi, topw, offs, cursor,
                                                     rowOf, wtOf, pos);
  moe_gather_kernel<<<NSLOT, 256, 0, stream>>>(rowOf, h2, hg);

  // 7. expert MLP (segmented GEMMs, only top-2 rows)
  gemm_bf16_kernel<EPI_GELU><<<dim3(HID / 128, NTOK / 128, NE), 256, 0, stream>>>(
      hg, w1, b1, nullptr, nullptr, he, nullptr, offs, counts,
      NTOK, DIM, HID);
  gemm_bf16_kernel<EPI_SCALE><<<dim3(DIM / 128, NTOK / 128, NE), 256, 0, stream>>>(
      he, w2, b2, nullptr, outg, nullptr, wtOf, offs, counts,
      NTOK, HID, DIM);

  // 8. residual + combine
  final_combine_kernel<<<dim3(NTOK, DIM / 256), 256, 0, stream>>>(x2, outg, pos, out);
}
